// Model_77653008711691
// MI455X (gfx1250) — matrix operand order repaired and checked by reading, not compiled, not run on hardware
//
#include <hip/hip_runtime.h>
#include <hip/hip_bf16.h>

// ---------------- model constants ----------------
#define Bn   32
#define Sn   512
#define Cn   64
#define PLn  16
#define Dn   128
#define PREDn 96
#define PNn  32
#define NFn  4096
#define ROWSn (Bn*Cn)   // 2048

typedef __attribute__((ext_vector_type(16))) __bf16        v16bf;
typedef __attribute__((ext_vector_type(2)))  __bf16        v2bf;
typedef __attribute__((ext_vector_type(8)))  float         v8f;
typedef __attribute__((ext_vector_type(8)))  unsigned int  v8u;

#define GEMM_FLAG_GELU 1
#define GEMM_FLAG_RES  2

// One-instruction f32x2 -> bf16x2 pack (RNE). Falls back to manual pack.
__device__ __forceinline__ unsigned cvt_pk_bf16(float a, float b) {
#if __has_builtin(__builtin_amdgcn_cvt_pk_bf16_f32)
  v2bf r = __builtin_amdgcn_cvt_pk_bf16_f32(a, b);
  return __builtin_bit_cast(unsigned, r);
#elif defined(__gfx1250__)
  unsigned r;
  asm("v_cvt_pk_bf16_f32 %0, %1, %2" : "=v"(r) : "v"(a), "v"(b));
  return r;
#else
  auto f2bf = [](float f) -> unsigned {
    unsigned u = __float_as_uint(f);
    u += 0x7fffu + ((u >> 16) & 1u);
    return u >> 16;
  };
  return f2bf(a) | (f2bf(b) << 16);
#endif
}
__device__ __forceinline__ float gelu_exact(float x) {
  return 0.5f * x * (1.0f + erff(x * 0.70710678118654752f));
}

// ---------------------------------------------------------------------------
// Generalized strided/batched bf16 WMMA GEMM:
//   C[r,n] = epilogue( sum_k A(r,k) * W(n,k) + bias[n] )
// A element  : A[aBatch*z + (r/G)*aOut + (r%G)*aIn + k*aK]
// W element  : W[z*N*K + n*K + k]            (torch Linear weight layout [out,in])
// bias       : bias[z*N + n]
// C element  : C[cBatch*z + (r/G)*cOut + (r%G)*cIn + n*cN]   (Res uses same map)
//
// Block tile 128(M) x 128(N), 8 waves of 32, wave tile 32x64 -> 8 WMMA/k-step.
// Software-pipelined: slab k+1 global float4 loads issue between the fragment
// ds_loads and the 8 v_wmma of slab k. AK1 template removes the strided-gather
// path from all contiguous-K GEMMs (everything except timecl).
// ---------------------------------------------------------------------------
template <bool AK1>
__global__ void __launch_bounds__(256)
gemm_bf16_wmma(const float* __restrict__ A, const float* __restrict__ W,
               const float* __restrict__ bias, const float* __restrict__ Res,
               const float* __restrict__ Pos, float* __restrict__ Cd,
               int M, int N, int K,
               int G, int aIn, long long aOut, int aK, long long aBatch,
               int cIn, long long cOut, int cN, long long cBatch,
               int posMod, int flags)
{
  __shared__ unsigned int Asu[128 * 16];   // 128 rows x 32 bf16 (16 dwords)
  __shared__ unsigned int Bsu[128 * 16];   // 128 rows x 32 bf16

  const int tid  = threadIdx.x;
  const int lane = tid & 31;
  const int wid  = tid >> 5;
  const int wmR  = wid >> 1;     // 0..3  -> wave M offset = wmR*32
  const int wmC  = wid & 1;      // 0..1  -> wave N offset = wmC*64
  const int z    = blockIdx.z;
  const int bm   = blockIdx.x * 128;
  const int bn   = blockIdx.y * 128;
  const float* __restrict__ Wz = W + (long long)z * N * K;
  const float* __restrict__ Az = A + (long long)z * aBatch;

  // ---- hoisted staging coordinates: thread stages 4 A rows and 4 B rows,
  //      always at k-quad kq = (tid&7)*4 within the 32-wide k-slab ----
  const int srow = tid >> 3;            // 0..31
  const int kq   = (tid & 7) * 4;
  long long aRow[4]; bool aVal[4];
  long long bRow[4]; bool bVal[4];
#pragma unroll
  for (int i = 0; i < 4; ++i) {
    int gr = bm + srow + 32 * i;
    aVal[i] = gr < M;
    int q = gr / G, rr = gr - q * G;
    aRow[i] = (long long)q * aOut + (long long)rr * aIn;
    int gn = bn + srow + 32 * i;
    bVal[i] = gn < N;
    bRow[i] = (long long)gn * K;
  }

  // slab loader: global -> registers (float4 per staged row)
  float4 aReg[4], bReg[4];
  auto loadSlab = [&](int k0) {
    const int gk = k0 + kq;
#pragma unroll
    for (int i = 0; i < 4; ++i) {
      float4 v = {0.f, 0.f, 0.f, 0.f};
      if (aVal[i]) {
        const float* p = Az + aRow[i];
        if (AK1) {
          if (gk + 4 <= K) v = *reinterpret_cast<const float4*>(p + gk);
          else {
            if (gk + 0 < K) v.x = p[gk + 0];
            if (gk + 1 < K) v.y = p[gk + 1];
            if (gk + 2 < K) v.z = p[gk + 2];
            if (gk + 3 < K) v.w = p[gk + 3];
          }
        } else {
          if (gk + 0 < K) v.x = p[(long long)(gk + 0) * aK];
          if (gk + 1 < K) v.y = p[(long long)(gk + 1) * aK];
          if (gk + 2 < K) v.z = p[(long long)(gk + 2) * aK];
          if (gk + 3 < K) v.w = p[(long long)(gk + 3) * aK];
        }
      }
      aReg[i] = v;
      float4 u = {0.f, 0.f, 0.f, 0.f};
      if (bVal[i]) {
        const float* p = Wz + bRow[i];
        if (gk + 4 <= K) u = *reinterpret_cast<const float4*>(p + gk);
        else {
          if (gk + 0 < K) u.x = p[gk + 0];
          if (gk + 1 < K) u.y = p[gk + 1];
          if (gk + 2 < K) u.z = p[gk + 2];
          if (gk + 3 < K) u.w = p[gk + 3];
        }
      }
      bReg[i] = u;
    }
  };

  v8f acc[2][4];
#pragma unroll
  for (int i = 0; i < 2; ++i)
#pragma unroll
    for (int j = 0; j < 4; ++j) acc[i][j] = 0.0f;

  const int mhalf = lane & 15;
  const int hs    = lane >> 4;
  const int kb    = hs ? 8 : 0;          // B dword base: K 16..31 for hi lanes

  loadSlab(0);                           // prologue prefetch

  for (int k0 = 0; k0 < K; k0 += 32) {
    // ---- commit prefetched slab to LDS (v_cvt_pk_bf16_f32, ds_store_b64) ----
#pragma unroll
    for (int i = 0; i < 4; ++i) {
      uint2 pa, pb;
      pa.x = cvt_pk_bf16(aReg[i].x, aReg[i].y);
      pa.y = cvt_pk_bf16(aReg[i].z, aReg[i].w);
      pb.x = cvt_pk_bf16(bReg[i].x, bReg[i].y);
      pb.y = cvt_pk_bf16(bReg[i].z, bReg[i].w);
      *reinterpret_cast<uint2*>(&Asu[(srow + 32 * i) * 16 + (kq >> 1)]) = pa;
      *reinterpret_cast<uint2*>(&Bsu[(srow + 32 * i) * 16 + (kq >> 1)]) = pb;
    }
    __syncthreads();

    // ---- fragment loads (A and B both in contiguous k = 16*half + s order) ----
    v8u afr[2], bfr[4];
#pragma unroll
    for (int t = 0; t < 2; ++t) {
      int mrow = wmR * 32 + t * 16 + mhalf;
#pragma unroll
      for (int j = 0; j < 8; ++j) {
        int kk = 2 * j + (hs ? 16 : 0);
        afr[t][j] = Asu[(mrow << 4) + (kk >> 1)];
      }
    }
#pragma unroll
    for (int t = 0; t < 4; ++t) {
      int ncol = wmC * 64 + t * 16 + mhalf;
#pragma unroll
      for (int j = 0; j < 8; ++j)
        bfr[t][j] = Bsu[(ncol << 4) + kb + j];
    }

    // ---- prefetch next slab: global loads overlap the WMMAs below ----
    if (k0 + 32 < K) loadSlab(k0 + 32);

    // ---- 8 WMMAs ----
#pragma unroll
    for (int ti = 0; ti < 2; ++ti)
#pragma unroll
      for (int tj = 0; tj < 4; ++tj)
        acc[ti][tj] = __builtin_amdgcn_wmma_f32_16x16x32_bf16(
            false, __builtin_bit_cast(v16bf, afr[ti]),
            false, __builtin_bit_cast(v16bf, bfr[tj]),
            (short)0, acc[ti][tj], false, false);
    __syncthreads();   // all frag reads done before next LDS commit
  }

  // ---- epilogue: bias (+pos) (+gelu) (+residual); row math hoisted ----
  const long long cB = (long long)z * cBatch;
  float biasv[4]; int gnv[4]; bool gnok[4];
#pragma unroll
  for (int tj = 0; tj < 4; ++tj) {
    int gn = bn + wmC * 64 + tj * 16 + mhalf;
    gnv[tj] = gn; gnok[tj] = gn < N;
    biasv[tj] = (bias && gn < N) ? bias[(long long)z * N + gn] : 0.0f;
  }
#pragma unroll
  for (int ti = 0; ti < 2; ++ti) {
#pragma unroll
    for (int i = 0; i < 8; ++i) {
      int gr = bm + wmR * 32 + ti * 16 + i + (hs << 3);
      if (gr < M) {
        int q = gr / G, rr = gr - q * G;
        long long rowCo = cB + (long long)q * cOut + (long long)rr * cIn;
        long long posRow = posMod ? (long long)(gr % posMod) * N : 0;
#pragma unroll
        for (int tj = 0; tj < 4; ++tj) {
          if (gnok[tj]) {
            float v = acc[ti][tj][i] + biasv[tj];
            if (posMod) v += Pos[posRow + gnv[tj]];
            if (flags & GEMM_FLAG_GELU) v = gelu_exact(v);
            long long co = rowCo + (long long)gnv[tj] * cN;
            if (flags & GEMM_FLAG_RES) v += Res[co];
            Cd[co] = v;
          }
        }
      }
    }
  }
}

// ---------------------------------------------------------------------------
// Row normalization: z = (x - mean) / (std_unbiased + 1e-4), x is [B,S,C].
// ---------------------------------------------------------------------------
__global__ void __launch_bounds__(256)
norm_kernel(const float* __restrict__ x, float* __restrict__ zn,
            float* __restrict__ mu, float* __restrict__ sdeps)
{
  __shared__ float red[256];
  __shared__ float sMean, sSd;
  int row = blockIdx.x;                 // b*C + c
  int b = row >> 6, c = row & 63;
  int tid = threadIdx.x;
  const float* xp = x + (long long)b * (Sn * Cn) + c;
  float v0 = xp[(long long)tid * Cn];
  float v1 = xp[(long long)(tid + 256) * Cn];
  red[tid] = v0 + v1;
  __syncthreads();
  for (int s = 128; s > 0; s >>= 1) { if (tid < s) red[tid] += red[tid + s]; __syncthreads(); }
  if (tid == 0) sMean = red[0] * (1.0f / 512.0f);
  __syncthreads();
  float m = sMean;
  float d0 = v0 - m, d1 = v1 - m;
  red[tid] = d0 * d0 + d1 * d1;
  __syncthreads();
  for (int s = 128; s > 0; s >>= 1) { if (tid < s) red[tid] += red[tid + s]; __syncthreads(); }
  if (tid == 0) { sSd = sqrtf(red[0] * (1.0f / 511.0f)) + 1e-4f; mu[row] = m; sdeps[row] = sSd; }
  __syncthreads();
  float sd = sSd;
  zn[(long long)row * Sn + tid]       = d0 / sd;
  zn[(long long)row * Sn + tid + 256] = d1 / sd;
}

// ---------------------------------------------------------------------------
// 512-pt radix-2 FFT in LDS.
// mode 0 (forward, real input): writes f_in patches [rows*PN, 32] = [re||im].
// mode 1 (inverse of h + i*h): writes cc row [ real(512) || imag(512) ] /512.
// ---------------------------------------------------------------------------
__global__ void __launch_bounds__(256)
fft512_kernel(const float* __restrict__ in, float* __restrict__ out, int mode)
{
  __shared__ float re[512], im[512];
  int row = blockIdx.x, tid = threadIdx.x;
  for (int t = tid; t < 512; t += 256) {
    int j = __brev((unsigned)t) >> 23;        // 9-bit reversal
    float v = in[(long long)row * 512 + t];
    re[j] = v;
    im[j] = mode ? v : 0.0f;
  }
  __syncthreads();
  float sgn = mode ? 1.0f : -1.0f;
  for (int len = 2; len <= 512; len <<= 1) {
    int half = len >> 1;
    int grp = tid / half, pos = tid % half;
    int i0 = grp * len + pos, i1 = i0 + half;
    float ang = sgn * 6.28318530717958647692f * (float)pos / (float)len;
    float s_, c_;
    __sincosf(ang, &s_, &c_);
    float br = re[i1], bi = im[i1];
    float tr = br * c_ - bi * s_;
    float ti = br * s_ + bi * c_;
    float ar = re[i0], ai = im[i0];
    re[i0] = ar + tr; im[i0] = ai + ti;
    re[i1] = ar - tr; im[i1] = ai - ti;
    __syncthreads();
  }
  if (mode == 0) {
    for (int t = tid; t < 512; t += 256) {
      int p = t >> 4, cix = t & 15;
      long long base = ((long long)row * PNn + p) * 32;
      out[base + cix]      = re[t];
      out[base + 16 + cix] = im[t];
    }
  } else {
    const float scl = 1.0f / 512.0f;
    for (int t = tid; t < 512; t += 256) {
      out[(long long)row * 1024 + t]       = re[t] * scl;
      out[(long long)row * 1024 + 512 + t] = im[t] * scl;
    }
  }
}

__global__ void __launch_bounds__(256)
concat_kernel(const float* __restrict__ a, const float* __restrict__ b, float* __restrict__ o)
{
  long long i = (long long)blockIdx.x * 256 + threadIdx.x;   // over 2048*1024
  int r = (int)(i >> 10), j = (int)(i & 1023);
  o[i] = (j < 512) ? a[(long long)r * 512 + j] : b[(long long)r * 512 + (j - 512)];
}

__global__ void __launch_bounds__(256)
final_kernel(const float* __restrict__ zout, const float* __restrict__ mu,
             const float* __restrict__ sde, float* __restrict__ out)
{
  int i = blockIdx.x * 256 + threadIdx.x;       // over B*PRED*C
  if (i >= Bn * PREDn * Cn) return;
  int c = i & 63;
  int p = (i >> 6) % PREDn;
  int b = i / (PREDn * Cn);
  int row = (b << 6) + c;
  out[i] = zout[(long long)row * PREDn + p] * sde[row] + mu[row];
}

// ---------------------------------------------------------------------------
extern "C" void kernel_launch(void* const* d_in, const int* in_sizes, int n_in,
                              void* d_out, int out_size, void* d_ws, size_t ws_size,
                              hipStream_t stream)
{
  (void)in_sizes; (void)n_in; (void)out_size; (void)ws_size;
  const float* x    = (const float*)d_in[0];
  // d_in[1] = mask (unused by the model)
  const float* sdW  = (const float*)d_in[2];  const float* sdb  = (const float*)d_in[3];
  const float* ipW  = (const float*)d_in[4];  const float* ipb  = (const float*)d_in[5];
  const float* pos  = (const float*)d_in[6];
  const float* t1W  = (const float*)d_in[7];  const float* t1b  = (const float*)d_in[8];
  const float* t2W  = (const float*)d_in[9];  const float* t2b  = (const float*)d_in[10];
  const float* c1W  = (const float*)d_in[11]; const float* c1b  = (const float*)d_in[12];
  const float* c2W  = (const float*)d_in[13]; const float* c2b  = (const float*)d_in[14];
  const float* p1W  = (const float*)d_in[15]; const float* p1b  = (const float*)d_in[16];
  const float* p2W  = (const float*)d_in[17]; const float* p2b  = (const float*)d_in[18];
  const float* ltsW = (const float*)d_in[19]; const float* ltsb = (const float*)d_in[20];
  const float* brW  = (const float*)d_in[21]; const float* brb  = (const float*)d_in[22];
  const float* h1W  = (const float*)d_in[23]; const float* h1b  = (const float*)d_in[24];
  const float* h2W  = (const float*)d_in[25]; const float* h2b  = (const float*)d_in[26];
  const float* h3W  = (const float*)d_in[27]; const float* h3b  = (const float*)d_in[28];
  const float* h4W  = (const float*)d_in[29]; const float* h4b  = (const float*)d_in[30];
  const float* dsW  = (const float*)d_in[31]; const float* dsb  = (const float*)d_in[32];
  const float* woW  = (const float*)d_in[33]; const float* wob  = (const float*)d_in[34];

  // ---- workspace layout (f32), total ~145 MB ----
  float* w     = (float*)d_ws;
  float* mu    = w;                         // 2048
  float* sde   = w + 2048;                  // 2048
  float* znorm = w + 4096;                  // 2048*512
  float* fin   = znorm + 1048576;           // 65536*32
  float* fpl   = fin + 2097152;             // 65536*16
  float* bufA  = fpl + 1048576;             // 2048*4096
  float* bufB  = bufA + 8388608;
  float* bufC  = bufB + 8388608;
  float* outs1 = bufC + 8388608;            // 2048*512
  float* hsum  = outs1 + 1048576;           // 2048*512
  float* cc    = hsum + 1048576;            // 2048*1024
  float* outf  = cc + 2097152;              // 2048*512
  float* cc2   = outf + 1048576;            // 2048*1024
  float* outs2 = cc2 + 2097152;             // 2048*512
  float* zout  = outs2 + 1048576;           // 2048*96

  auto gemm = [&](const float* A, const float* W, const float* bias,
                  const float* Res, const float* Pos, float* Cd,
                  int M, int N, int K,
                  int G, int aIn, long long aOut, int aK, long long aBatch,
                  int cIn, long long cOut, int cN, long long cBatch,
                  int batches, int posMod, int flags) {
    dim3 g((M + 127) / 128, (N + 127) / 128, batches), blk(256);
    if (aK == 1)
      hipLaunchKernelGGL((gemm_bf16_wmma<true>), g, blk, 0, stream, A, W, bias, Res, Pos, Cd,
                         M, N, K, G, aIn, aOut, aK, aBatch, cIn, cOut, cN, cBatch,
                         posMod, flags);
    else
      hipLaunchKernelGGL((gemm_bf16_wmma<false>), g, blk, 0, stream, A, W, bias, Res, Pos, Cd,
                         M, N, K, G, aIn, aOut, aK, aBatch, cIn, cOut, cN, cBatch,
                         posMod, flags);
  };
  // one mixer sublayer: o = (gelu(u@W1^T + b1))@W2^T + b2 + u, strided/batched
  auto sub = [&](float* u, float* t, float* o,
                 const float* W1, const float* b1, const float* W2, const float* b2,
                 int kind) {
    if (kind == 0) {        // timecl: batch=D, M=2048(rows b,c), N=K=PN
      gemm(u, W1, b1, nullptr, nullptr, t, 2048, 32, 32,
           2048, 4096, 0, 128, 1, 4096, 0, 128, 1, 128, 0, GEMM_FLAG_GELU);
      gemm(t, W2, b2, u, nullptr, o, 2048, 32, 32,
           2048, 4096, 0, 128, 1, 4096, 0, 128, 1, 128, 0, GEMM_FLAG_RES);
    } else if (kind == 1) { // channelcl: batch=C, M=B*PN, N=K=D
      gemm(u, W1, b1, nullptr, nullptr, t, 1024, 128, 128,
           32, 128, 262144, 1, 4096, 128, 262144, 1, 4096, 64, 0, GEMM_FLAG_GELU);
      gemm(t, W2, b2, u, nullptr, o, 1024, 128, 128,
           32, 128, 262144, 1, 4096, 128, 262144, 1, 4096, 64, 0, GEMM_FLAG_RES);
    } else {                // patchnumcl: batch=PN, M=B*C, N=K=D
      gemm(u, W1, b1, nullptr, nullptr, t, 2048, 128, 128,
           2048, 4096, 0, 1, 128, 4096, 0, 1, 128, 32, 0, GEMM_FLAG_GELU);
      gemm(t, W2, b2, u, nullptr, o, 2048, 128, 128,
           2048, 4096, 0, 1, 128, 4096, 0, 1, 128, 32, 0, GEMM_FLAG_RES);
    }
  };
  auto runMix = [&](float* A, float* Bb, float* Cc) -> float* {
    sub(A,  Bb, Cc, t1W, t1b, t2W, t2b, 0);   // out in Cc
    sub(Cc, A,  Bb, c1W, c1b, c2W, c2b, 1);   // out in Bb
    sub(Bb, A,  Cc, p1W, p1b, p2W, p2b, 2);   // out in Cc
    return Cc;
  };

  // ---- stage 0: normalize + forward FFT ----
  hipLaunchKernelGGL(norm_kernel,   dim3(ROWSn), dim3(256), 0, stream, x, znorm, mu, sde);
  hipLaunchKernelGGL(fft512_kernel, dim3(ROWSn), dim3(256), 0, stream, znorm, fin, 0);

  // seqdivide: f = [re||im] @ sd_W^T + sd_b   (M=65536, N=16, K=32)
  gemm(fin, sdW, sdb, nullptr, nullptr, fpl, 65536, 16, 32,
       65536, 32, 0, 1, 0, 16, 0, 1, 0, 1, 0, 0);

  // ---- z path: embed (+pos), mix, lineartoseq ----
  gemm(znorm, ipW, ipb, nullptr, pos, bufA, 65536, 128, 16,
       65536, 16, 0, 1, 0, 128, 0, 1, 0, 1, PNn, 0);
  float* zm = runMix(bufA, bufB, bufC);
  gemm(zm, ltsW, ltsb, nullptr, nullptr, outs1, 2048, 512, 4096,
       2048, 4096, 0, 1, 0, 512, 0, 1, 0, 1, 0, 0);

  // ---- f path: embed (+pos), mix, breakn ----
  gemm(fpl, ipW, ipb, nullptr, pos, bufA, 65536, 128, 16,
       65536, 16, 0, 1, 0, 128, 0, 1, 0, 1, PNn, 0);
  float* fm = runMix(bufA, bufB, bufC);
  gemm(fm, brW, brb, nullptr, nullptr, bufA, 65536, 128, 128,
       65536, 128, 0, 1, 0, 128, 0, 1, 0, 1, 0, 0);

  // ---- Flatten head: h = gelu(h@W+b)+h  x3, then h4 ----
  gemm(bufA, h1W, h1b, bufA, nullptr, bufB, 2048, 4096, 4096,
       2048, 4096, 0, 1, 0, 4096, 0, 1, 0, 1, 0, GEMM_FLAG_GELU | GEMM_FLAG_RES);
  gemm(bufB, h2W, h2b, bufB, nullptr, bufA, 2048, 4096, 4096,
       2048, 4096, 0, 1, 0, 4096, 0, 1, 0, 1, 0, GEMM_FLAG_GELU | GEMM_FLAG_RES);
  gemm(bufA, h3W, h3b, bufA, nullptr, bufB, 2048, 4096, 4096,
       2048, 4096, 0, 1, 0, 4096, 0, 1, 0, 1, 0, GEMM_FLAG_GELU | GEMM_FLAG_RES);
  gemm(bufB, h4W, h4b, nullptr, nullptr, hsum, 2048, 512, 4096,
       2048, 4096, 0, 1, 0, 512, 0, 1, 0, 1, 0, 0);

  // ---- ifft(h + i*h) -> [real || imag], then two dsdivide, then W_out ----
  hipLaunchKernelGGL(fft512_kernel, dim3(ROWSn), dim3(256), 0, stream, hsum, cc, 1);
  gemm(cc, dsW, dsb, nullptr, nullptr, outf, 2048, 512, 1024,
       2048, 1024, 0, 1, 0, 512, 0, 1, 0, 1, 0, 0);
  hipLaunchKernelGGL(concat_kernel, dim3((ROWSn * 1024) / 256), dim3(256), 0, stream,
                     outs1, outf, cc2);
  gemm(cc2, dsW, dsb, nullptr, nullptr, outs2, 2048, 512, 1024,
       2048, 1024, 0, 1, 0, 512, 0, 1, 0, 1, 0, 0);
  gemm(outs2, woW, wob, nullptr, nullptr, zout, 2048, 96, 512,
       2048, 512, 0, 1, 0, 96, 0, 1, 0, 1, 0, 0);

  // ---- de-normalize + transpose to [B, PRED, C] ----
  hipLaunchKernelGGL(final_kernel, dim3((Bn * PREDn * Cn + 255) / 256), dim3(256), 0, stream,
                     zout, mu, sde, (float*)d_out);
}
